// SimpleMNISTModel_24730421690978
// MI455X (gfx1250) — matrix operand order repaired and checked by reading, not compiled, not run on hardware
//
#include <hip/hip_runtime.h>
#include <hip/hip_bf16.h>

// ---- model constants (from reference) ----
#define H_HARM  32
#define PH_D0   7.0f
#define HID     128
#define NEXP    4
#define EDIM    64
#define NCLS    3
#define IN_DIM  784
#define ENH     192

// ---- tiling ----
#define BT      128          // batch rows per block
#define KC      128          // K chunk for GEMM1 (6 full chunks + 16-wide tail)

// ---- vector types ----
typedef __attribute__((ext_vector_type(16))) _Float16 v16h;
typedef __attribute__((ext_vector_type(8)))  _Float16 v8h;
typedef __attribute__((ext_vector_type(8)))  float    v8f;
typedef __attribute__((ext_vector_type(4)))  float    v4f;

// ---- LDS arena (byte offsets; phase-aliased) ----
// phase A : xk 128x808 f16 (206848 B) @0 ; wk0 @206848 ; wk1 @241664 (34816 B each)
// phase A end/B: proj 128x132 f32 (67584 B) @206848 (aliases wk0/wk1)
// phase B/C: enh 128x200 f16 (51200) @0 ; gate 128x4 f32 @51200 ; we2 256x200 f16 @53248
// phase D : cmb 128x72 f16 @0 ; wct 128x72 f16 @18432 ; ctx 128x132 f32 @53248
#define XK_OFF     0u
#define WK0_OFF    206848u
#define WK1_OFF    241664u
#define PROJ_OFF   206848u
#define ENH_OFF    0u
#define GATE_OFF   51200u
#define WE2_OFF    53248u
#define CMB_OFF    0u
#define WCT_OFF    18432u
#define CTX_OFF    53248u
#define SMEM_BYTES 276480u

#define XK_S   808   // half-elem strides; rows 16B-aligned; bank-shifted
#define WK_S   136
#define ENH_S  200
#define WE_S   200
#define PRJ_S  132
#define CMB_S  72
#define WCT_S  72
#define CTX_S  132

// ---- WMMA wrapper (gfx1250 only; host pass gets a stub) ----
__device__ __forceinline__ v8f wmma16(v16h a, v16h b, v8f c) {
#if defined(__gfx1250__)
  return __builtin_amdgcn_wmma_f32_16x16x32_f16(false, a, false, b, (short)0, c,
                                                false, false);
#else
  (void)a; (void)b;
  return c;
#endif
}

// A-fragment (16x32 f16): lanes 0-15 K=base+0..7 / +16..23, lanes 16-31 +8.
__device__ __forceinline__ v16h load_a(const _Float16* rowbase, int kb, int lane) {
  const _Float16* p = rowbase + kb + ((lane >> 4) << 3);
  v8h lo = *(const v8h*)(p);
  v8h hi = *(const v8h*)(p + 16);
  return __builtin_shufflevector(lo, hi, 0,1,2,3,4,5,6,7,8,9,10,11,12,13,14,15);
}

// B-fragment (32x16 f16): lanes 0-15 K=0..7 / 16..23, lanes 16-31 +8 (col-major LDS).
__device__ __forceinline__ v16h load_b(const _Float16* colbase, int kb, int lane) {
  const _Float16* p = colbase + kb + ((lane >> 4) << 3);
  v8h lo = *(const v8h*)(p);
  v8h hi = *(const v8h*)(p + 16);
  return __builtin_shufflevector(lo, hi, 0,1,2,3,4,5,6,7,8,9,10,11,12,13,14,15);
}

extern "C" __global__ __launch_bounds__(256)
void phasor_moe_fused(const float* __restrict__ x,
                      const float* __restrict__ W_in,  const float* __restrict__ b_in,
                      const float* __restrict__ W_gate,const float* __restrict__ b_gate,
                      const float* __restrict__ W_exp, const float* __restrict__ b_exp,
                      const float* __restrict__ W_ctx, const float* __restrict__ b_ctx,
                      const float* __restrict__ W_out, const float* __restrict__ b_out,
                      float* __restrict__ out)
{
  extern __shared__ __align__(16) char smem[];
  const int tid  = threadIdx.x;
  const int lane = tid & 31;
  const int wave = tid >> 5;
  const int l15  = lane & 15;
  const int lhi8 = (lane >> 4) << 3;       // C/D layout: rows +8 for lanes 16-31
  const int rB   = blockIdx.x * BT;

  // wave tiling for 128-wide output stages: 4 row-groups x 2 col-groups
  const int wr = wave >> 1;                // 0..3 -> 32 rows
  const int wc = wave & 1;                 // 0..1 -> 64 cols

  _Float16* xk  = (_Float16*)(smem + XK_OFF);
  _Float16* wk0 = (_Float16*)(smem + WK0_OFF);
  _Float16* wk1 = (_Float16*)(smem + WK1_OFF);

  // ============ Phase A: projected = x @ W_in + b_in (K=784) ============
  // 1) whole x tile -> f16 LDS in one deep pipelined burst; non-temporal so the
  //    205MB x stream never evicts the L2-resident weights.
#pragma unroll 2
  for (int it = 0; it < 98; ++it) {        // 128 rows x 196 float4
    int idx = tid + it * 256;
    int r   = idx / 196;
    int c4  = idx - r * 196;
    v4f xv = __builtin_nontemporal_load(
        (const v4f*)(x + (size_t)(rB + r) * IN_DIM + c4 * 4));
    _Float16* d = xk + r * XK_S + c4 * 4;
    d[0] = (_Float16)xv.x; d[1] = (_Float16)xv.y;
    d[2] = (_Float16)xv.z; d[3] = (_Float16)xv.w;
  }
#pragma unroll
  for (int it = 0; it < 12; ++it) {        // zero-pad k in [784, 808)
    int idx = tid + it * 256;              // 128 rows x 24 halves
    int r = idx / 24;
    int c = idx - r * 24;
    xk[r * XK_S + IN_DIM + c] = (_Float16)0.f;
  }
  // 2) W chunk 0 -> wk0 (col-major f16); float4 loads, paired b32 stores
#pragma unroll 2
  for (int it = 0; it < 8; ++it) {
    int idx = tid + it * 256;              // 64 k-pairs x 32 n-quads
    int kp = idx >> 5, n4 = idx & 31;
    v4f w0 = *(const v4f*)(W_in + (size_t)(2 * kp)     * HID + n4 * 4);
    v4f w1 = *(const v4f*)(W_in + (size_t)(2 * kp + 1) * HID + n4 * 4);
#pragma unroll
    for (int j = 0; j < 4; ++j) {
      _Float16* p = wk0 + (n4 * 4 + j) * WK_S + kp * 2;
      p[0] = (_Float16)w0[j]; p[1] = (_Float16)w1[j];
    }
  }
  __syncthreads();

  v8f acc[2][4] = {};
  for (int ch = 0; ch < 6; ++ch) {
    _Float16* cur = (ch & 1) ? wk1 : wk0;
    _Float16* nxt = (ch & 1) ? wk0 : wk1;
    // issue next W chunk's global loads before compute (L2 latency hides under WMMA)
    v4f wrg0[8], wrg1[8];
    const int kn0 = (ch + 1) * KC;
#pragma unroll
    for (int it = 0; it < 8; ++it) {
      int idx = tid + it * 256;
      int kp = idx >> 5, n4 = idx & 31;
      int k0g = kn0 + kp * 2;
      v4f z = {0.f, 0.f, 0.f, 0.f};
      wrg0[it] = (k0g     < IN_DIM)
                   ? *(const v4f*)(W_in + (size_t)k0g       * HID + n4 * 4) : z;
      wrg1[it] = (k0g + 1 < IN_DIM)
                   ? *(const v4f*)(W_in + (size_t)(k0g + 1) * HID + n4 * 4) : z;
    }
    // compute current chunk: 4 k-steps x 8 WMMA
#pragma unroll
    for (int ks = 0; ks < 4; ++ks) {
      const int kg = ch * KC + ks * 32;    // offset in resident x tile
      const int kl = ks * 32;              // offset in W chunk buffer
      v16h a0 = load_a(xk + (wr * 32 +      l15) * XK_S, kg, lane);
      v16h a1 = load_a(xk + (wr * 32 + 16 + l15) * XK_S, kg, lane);
      v16h bf[4];
#pragma unroll
      for (int ct = 0; ct < 4; ++ct)
        bf[ct] = load_b(cur + (wc * 64 + ct * 16 + l15) * WK_S, kl, lane);
#pragma unroll
      for (int ct = 0; ct < 4; ++ct) {
        acc[0][ct] = wmma16(a0, bf[ct], acc[0][ct]);
        acc[1][ct] = wmma16(a1, bf[ct], acc[1][ct]);
      }
    }
    // convert + store next chunk into the other buffer (paired b32 stores)
#pragma unroll
    for (int it = 0; it < 8; ++it) {
      int idx = tid + it * 256;
      int kp = idx >> 5, n4 = idx & 31;
#pragma unroll
      for (int j = 0; j < 4; ++j) {
        _Float16* p = nxt + (n4 * 4 + j) * WK_S + kp * 2;
        p[0] = (_Float16)wrg0[it][j];
        p[1] = (_Float16)wrg1[it][j];
      }
    }
    __syncthreads();
  }
  {  // tail chunk 6: k = 768..799 (zero-padded beyond 783), one k-step, in wk0
    const int kg = 6 * KC;
    v16h a0 = load_a(xk + (wr * 32 +      l15) * XK_S, kg, lane);
    v16h a1 = load_a(xk + (wr * 32 + 16 + l15) * XK_S, kg, lane);
    v16h bf[4];
#pragma unroll
    for (int ct = 0; ct < 4; ++ct)
      bf[ct] = load_b(wk0 + (wc * 64 + ct * 16 + l15) * WK_S, 0, lane);
#pragma unroll
    for (int ct = 0; ct < 4; ++ct) {
      acc[0][ct] = wmma16(a0, bf[ct], acc[0][ct]);
      acc[1][ct] = wmma16(a1, bf[ct], acc[1][ct]);
    }
  }
  __syncthreads();                         // wk buffers dead; proj aliases them

  float* proj = (float*)(smem + PROJ_OFF);
#pragma unroll
  for (int rt = 0; rt < 2; ++rt)
#pragma unroll
    for (int ct = 0; ct < 4; ++ct) {
      int col = wc * 64 + ct * 16 + l15;
      float bi = b_in[col];
#pragma unroll
      for (int i = 0; i < 8; ++i) {
        int row = wr * 32 + rt * 16 + lhi8 + i;
        proj[row * PRJ_S + col] = acc[rt][ct][i] + bi;
      }
    }
  __syncthreads();

  // ====== Phase B: mean -> phasor bank -> enhanced(f16) -> softmax gate ======
  _Float16* enh  = (_Float16*)(smem + ENH_OFF);
  float*    gate = (float*)   (smem + GATE_OFF);
  if (tid < BT) {
    int r = tid;
    float s = 0.f;
    for (int h = 0; h < HID; ++h) s += proj[r * PRJ_S + h];
    float t  = s * (1.0f / HID);
    float c1 = cosf(t * PH_D0), s1 = sinf(t * PH_D0);
    float ck = c1, sk = s1;                // harmonic k via rotation recurrence
#pragma unroll
    for (int hh = 0; hh < H_HARM; ++hh) {
      enh[r * ENH_S + HID + hh]          = (_Float16)ck;
      enh[r * ENH_S + HID + H_HARM + hh] = (_Float16)sk;
      float cn = ck * c1 - sk * s1;
      float sn = sk * c1 + ck * s1;
      ck = cn; sk = sn;
    }
  }
#pragma unroll 4
  for (int it = 0; it < 64; ++it) {        // projected -> enhanced[:,0:128]
    int idx = tid + it * 256;
    int r = idx >> 7, h = idx & 127;
    enh[r * ENH_S + h] = (_Float16)proj[r * PRJ_S + h];
  }
  __syncthreads();
  if (tid < BT) {                          // gate = softmax(enh @ W_gate + b_gate)
    int r = tid;
    float g0 = b_gate[0], g1 = b_gate[1], g2 = b_gate[2], g3 = b_gate[3];
    for (int j = 0; j < ENH; ++j) {
      float v = (float)enh[r * ENH_S + j];
      const float* wg = W_gate + j * NEXP;
      g0 += v * wg[0]; g1 += v * wg[1]; g2 += v * wg[2]; g3 += v * wg[3];
    }
    float m = fmaxf(fmaxf(g0, g1), fmaxf(g2, g3));
    g0 = expf(g0 - m); g1 = expf(g1 - m); g2 = expf(g2 - m); g3 = expf(g3 - m);
    float inv = 1.0f / (g0 + g1 + g2 + g3);
    gate[r * 4 + 0] = g0 * inv; gate[r * 4 + 1] = g1 * inv;
    gate[r * 4 + 2] = g2 * inv; gate[r * 4 + 3] = g3 * inv;
  }
  // all experts' weights -> f16 col-major [256][200]
  _Float16* we2 = (_Float16*)(smem + WE2_OFF);
  for (int kk = 0; kk < ENH; ++kk) {
    int c = tid;                           // 256 threads = 256 expert columns
    int e = c >> 6, d = c & 63;
    we2[c * WE_S + kk] = (_Float16)W_exp[((size_t)e * ENH + kk) * EDIM + d];
  }
  __syncthreads();

  // ====== Phase C: experts GEMM + relu + gate combine (registers only) ======
  v8f comb[2][2] = {};
  for (int e = 0; e < NEXP; ++e) {
    v8f ea[2][2] = {};
#pragma unroll
    for (int ks = 0; ks < 6; ++ks) {       // K = 192
      const int kb = ks * 32;
      v16h a0 = load_a(enh + (wr * 32 +      l15) * ENH_S, kb, lane);
      v16h a1 = load_a(enh + (wr * 32 + 16 + l15) * ENH_S, kb, lane);
      v16h b0 = load_b(we2 + (e * EDIM + wc * 32 +      l15) * WE_S, kb, lane);
      v16h b1 = load_b(we2 + (e * EDIM + wc * 32 + 16 + l15) * WE_S, kb, lane);
      ea[0][0] = wmma16(a0, b0, ea[0][0]);
      ea[0][1] = wmma16(a0, b1, ea[0][1]);
      ea[1][0] = wmma16(a1, b0, ea[1][0]);
      ea[1][1] = wmma16(a1, b1, ea[1][1]);
    }
#pragma unroll
    for (int rt = 0; rt < 2; ++rt)
#pragma unroll
      for (int ct = 0; ct < 2; ++ct) {
        int col = wc * 32 + ct * 16 + l15;
        float be = b_exp[e * EDIM + col];
#pragma unroll
        for (int i = 0; i < 8; ++i) {
          int row = wr * 32 + rt * 16 + lhi8 + i;
          float v = fmaxf(ea[rt][ct][i] + be, 0.f);
          comb[rt][ct][i] += gate[row * 4 + e] * v;   // same lane owns (row,col) per e
        }
      }
  }
  __syncthreads();                         // enh/we2/gate dead after this point

  // combined -> f16 [128][72]; W_ctx -> f16 col-major [128][72]
  _Float16* cmb = (_Float16*)(smem + CMB_OFF);
  _Float16* wct = (_Float16*)(smem + WCT_OFF);
#pragma unroll
  for (int rt = 0; rt < 2; ++rt)
#pragma unroll
    for (int ct = 0; ct < 2; ++ct)
#pragma unroll
      for (int i = 0; i < 8; ++i) {
        int row = wr * 32 + rt * 16 + lhi8 + i;
        int col = wc * 32 + ct * 16 + l15;
        cmb[row * CMB_S + col] = (_Float16)comb[rt][ct][i];
      }
#pragma unroll 4
  for (int it = 0; it < 32; ++it) {
    int idx = tid + it * 256;              // 64x128 elements
    int k = idx >> 7, n = idx & 127;
    wct[n * WCT_S + k] = (_Float16)W_ctx[(size_t)k * HID + n];
  }
  __syncthreads();

  // ====== Phase D: context = tanh(comb @ W_ctx + b_ctx); logits ======
  float* ctx = (float*)(smem + CTX_OFF);
  {
    v8f cacc[2][4] = {};
#pragma unroll
    for (int ks = 0; ks < 2; ++ks) {       // K = 64
      const int kb = ks * 32;
      v16h a0 = load_a(cmb + (wr * 32 +      l15) * CMB_S, kb, lane);
      v16h a1 = load_a(cmb + (wr * 32 + 16 + l15) * CMB_S, kb, lane);
      v16h bf[4];
#pragma unroll
      for (int ct = 0; ct < 4; ++ct)
        bf[ct] = load_b(wct + (wc * 64 + ct * 16 + l15) * WCT_S, kb, lane);
#pragma unroll
      for (int ct = 0; ct < 4; ++ct) {
        cacc[0][ct] = wmma16(a0, bf[ct], cacc[0][ct]);
        cacc[1][ct] = wmma16(a1, bf[ct], cacc[1][ct]);
      }
    }
#pragma unroll
    for (int rt = 0; rt < 2; ++rt)
#pragma unroll
      for (int ct = 0; ct < 4; ++ct) {
        int col = wc * 64 + ct * 16 + l15;
        float bc = b_ctx[col];
#pragma unroll
        for (int i = 0; i < 8; ++i) {
          int row = wr * 32 + rt * 16 + lhi8 + i;
          ctx[row * CTX_S + col] = tanhf(cacc[rt][ct][i] + bc);
        }
      }
  }
  __syncthreads();
  if (tid < BT) {                          // logits = ctx @ W_out + b_out
    int r = tid;
    float l0 = b_out[0], l1 = b_out[1], l2 = b_out[2];
    for (int h = 0; h < HID; ++h) {
      float c = ctx[r * CTX_S + h];
      const float* wo = W_out + h * NCLS;
      l0 += c * wo[0]; l1 += c * wo[1]; l2 += c * wo[2];
    }
    float* o = out + (size_t)(rB + r) * NCLS;
    __builtin_nontemporal_store(l0, o + 0);
    __builtin_nontemporal_store(l1, o + 1);
    __builtin_nontemporal_store(l2, o + 2);
  }
}

extern "C" void kernel_launch(void* const* d_in, const int* in_sizes, int n_in,
                              void* d_out, int out_size, void* d_ws, size_t ws_size,
                              hipStream_t stream) {
  (void)n_in; (void)out_size; (void)d_ws; (void)ws_size;
  const float* x      = (const float*)d_in[0];
  const float* W_in   = (const float*)d_in[1];
  const float* b_in   = (const float*)d_in[2];
  const float* W_gate = (const float*)d_in[3];
  const float* b_gate = (const float*)d_in[4];
  const float* W_exp  = (const float*)d_in[5];
  const float* b_exp  = (const float*)d_in[6];
  const float* W_ctx  = (const float*)d_in[7];
  const float* b_ctx  = (const float*)d_in[8];
  const float* W_out  = (const float*)d_in[9];
  const float* b_out  = (const float*)d_in[10];
  float* out = (float*)d_out;

  const int B = in_sizes[0] / IN_DIM;      // 65536
  const int blocks = B / BT;               // 512

  (void)hipFuncSetAttribute(reinterpret_cast<const void*>(phasor_moe_fused),
                            hipFuncAttributeMaxDynamicSharedMemorySize,
                            (int)SMEM_BYTES);
  phasor_moe_fused<<<blocks, 256, SMEM_BYTES, stream>>>(
      x, W_in, b_in, W_gate, b_gate, W_exp, b_exp, W_ctx, b_ctx, W_out, b_out, out);
}
